// TromptCell_86045374808938
// MI455X (gfx1250) — matrix operand order repaired and checked by reading, not compiled, not run on hardware
//
#include <hip/hip_runtime.h>

#define DEV __device__ __forceinline__

typedef _Float16 half8 __attribute__((ext_vector_type(8)));
typedef _Float16 v16h  __attribute__((ext_vector_type(16)));
typedef float    v8f   __attribute__((ext_vector_type(8)));

constexpr int B_ = 2048, C_ = 100, P_ = 128, D_ = 192;
constexpr int S_PREV = 200;   // LDS row stride in halves (padded vs 192 for banking)
constexpr int S_XP   = 200;
constexpr int S_MASK = 136;   // 128 padded-C + 8
constexpr int S_XE   = 136;

// workspace layout (bytes):
//  ws_col : f16 [128][192] at 0        (49152)  = LN(emb_column), rows >=100 zero (B^T for GEMM2)
//  ws_wt  : f16 [192][192] at 49152    (73728)  = W_imp[(D+k)*D+n] transposed -> [n][k] (B^T for GEMM1)
//  ws_pc  : f32 [128][192] at 122880   (98304)  = LN(emb_prompt)@W_imp_top + b_imp + emb_prompt
constexpr size_t WS_COL = 0, WS_WT = 49152, WS_PC = 49152 + 73728;

DEV v16h make16(half8 lo, half8 hi) {
  return __builtin_shufflevector(lo, hi, 0,1,2,3,4,5,6,7,8,9,10,11,12,13,14,15);
}

// A fragment (16x32 f16, row-major [M][K] source). Per ISA: lanes 0-15 rows m, K = {0..7,16..23};
// lanes 16-31 same rows, K = {8..15,24..31}.
DEV v16h ldA(const _Float16* base, int strideH, int mt, int k0, int lane) {
  int m  = mt * 16 + (lane & 15);
  int kb = k0 + ((lane & 16) ? 8 : 0);
  const _Float16* p = base + m * strideH + kb;
  half8 lo = *(const half8*)(p);
  half8 hi = *(const half8*)(p + 16);
  return make16(lo, hi);
}

// B fragment from B^T storage [N][K]. Lanes 0-15: col n, K = {0..7,16..23}; lanes 16-31: col n, K = {8..15,24..31}.
DEV v16h ldBT(const _Float16* bt, int strideH, int nt, int k0, int lane) {
  int n  = nt * 16 + (lane & 15);
  int kb = k0 + ((lane & 16) ? 8 : 0);
  const _Float16* p = bt + n * strideH + kb;
  half8 lo = *(const half8*)(p);
  half8 hi = *(const half8*)(p + 16);
  return make16(lo, hi);
}

DEV v8f wmma16(v16h a, v16h b, v8f c) {
  return __builtin_amdgcn_wmma_f32_16x16x32_f16(false, a, false, b, (short)0, c, false, false);
}

// ---------------- prep kernels ----------------

__global__ void k_colln(const float* emb_col, const float* w, const float* b, _Float16* ws_col) {
  int c = threadIdx.x;
  if (c >= 128) return;
  if (c < C_) {
    const float* row = emb_col + c * D_;
    float s = 0.f, ss = 0.f;
    for (int d = 0; d < D_; ++d) { float v = row[d]; s += v; ss += v * v; }
    float m = s / D_, var = ss / D_ - m * m;
    float rs = rsqrtf(var + 1e-5f);
    for (int d = 0; d < D_; ++d)
      ws_col[c * D_ + d] = (_Float16)((row[d] - m) * rs * w[d] + b[d]);
  } else {
    for (int d = 0; d < D_; ++d) ws_col[c * D_ + d] = (_Float16)0.f;
  }
}

__global__ void k_wt(const float* W_imp, _Float16* ws_wt) {
  int i = blockIdx.x * blockDim.x + threadIdx.x;
  if (i >= D_ * D_) return;
  int n = i / D_, k = i % D_;
  ws_wt[n * D_ + k] = (_Float16)W_imp[(D_ + k) * D_ + n];
}

__global__ void k_pconst(const float* embp, const float* lnw, const float* lnb,
                         const float* W_imp, const float* b_imp, float* ws_pc) {
  int p = blockIdx.x;      // 0..127
  int d = threadIdx.x;     // 0..191
  const float* row = embp + p * D_;
  float s = 0.f, ss = 0.f;
  for (int e = 0; e < D_; ++e) { float v = row[e]; s += v; ss += v * v; }
  float m = s / D_, var = ss / D_ - m * m;
  float rs = rsqrtf(var + 1e-5f);
  float acc = 0.f;
  for (int e = 0; e < D_; ++e) {
    float lnp = (row[e] - m) * rs * lnw[e] + lnb[e];
    acc += lnp * W_imp[e * D_ + d];
  }
  ws_pc[p * D_ + d] = acc + b_imp[d] + row[d];
}

// ---------------- main fused kernel: one block per batch element ----------------

__global__ void __launch_bounds__(256)
k_main(const float* __restrict__ x, const float* __restrict__ prev,
       const float* __restrict__ Wfe, const float* __restrict__ bfe,
       const float* __restrict__ lnew, const float* __restrict__ lneb,
       const _Float16* __restrict__ ws_col, const _Float16* __restrict__ ws_wt,
       const float* __restrict__ ws_pc,
       const float* __restrict__ Wexp, const float* __restrict__ bexp,
       float* __restrict__ out) {
  // slot0: prev f16 [128][200] (51200 B)  -> reused as mask f16 [128][136] (34816 B)
  // slot1: x_prompt f16 [128][200] (51200) -> reused as x_emb^T f16 [192][136] (52224 B)
  // tail : LN stats scratch (3072 B)
  __shared__ __align__(16) char smem[51200 + 52224 + 3072];
  _Float16* prevL = (_Float16*)smem;
  _Float16* maskL = (_Float16*)smem;
  _Float16* xpL   = (_Float16*)(smem + 51200);
  _Float16* xeL   = (_Float16*)(smem + 51200);
  float* spart  = (float*)(smem + 103424);        // [256]
  float* sspart = (float*)(smem + 103424 + 1024); // [256]
  float* mrow   = (float*)(smem + 103424 + 2048); // [128]
  float* rsrow  = (float*)(smem + 103424 + 2560); // [128]

  const int tid = threadIdx.x, lane = tid & 31, wave = tid >> 5;
  const int b = blockIdx.x;
  const int mt = wave;                         // 8 waves -> 8 row tiles of 16
  const int n16 = (lane & 15);
  const int mr  = mt * 16 + ((lane & 16) ? 8 : 0);  // C/D fragment row base

  // ---- stage prev[b] f32 -> f16 LDS (coalesced float4) ----
  const float4* pv = (const float4*)(prev + (size_t)b * P_ * D_);
  for (int g = tid; g < P_ * D_ / 4; g += 256) {
    float4 v = pv[g];
    int row = g / (D_ / 4), col = (g % (D_ / 4)) * 4;
    _Float16* dst = prevL + row * S_PREV + col;
    dst[0] = (_Float16)v.x; dst[1] = (_Float16)v.y;
    dst[2] = (_Float16)v.z; dst[3] = (_Float16)v.w;
  }
  __syncthreads();

  // ---- GEMM1: x_prompt = prev @ W_bot + pconst  (M=128,N=192,K=192) ----
  v16h a1[6];
  #pragma unroll
  for (int kk = 0; kk < 6; ++kk) a1[kk] = ldA(prevL, S_PREV, mt, kk * 32, lane);
  for (int nt = 0; nt < 12; ++nt) {
    int n = nt * 16 + n16;
    v8f acc;
    #pragma unroll
    for (int r = 0; r < 8; ++r) acc[r] = ws_pc[(mr + r) * D_ + n];
    #pragma unroll
    for (int kk = 0; kk < 6; ++kk)
      acc = wmma16(a1[kk], ldBT(ws_wt, D_, nt, kk * 32, lane), acc);
    #pragma unroll
    for (int r = 0; r < 8; ++r) xpL[(mr + r) * S_XP + n] = (_Float16)acc[r];
  }
  __syncthreads();

  // ---- GEMM2: logits = x_prompt @ colLN^T  (N=112 padded, K=192), softmax in registers ----
  v16h a2[6];
  #pragma unroll
  for (int kk = 0; kk < 6; ++kk) a2[kk] = ldA(xpL, S_XP, mt, kk * 32, lane);
  v8f lg[7];
  #pragma unroll
  for (int nt = 0; nt < 7; ++nt) {
    v8f acc = {};
    #pragma unroll
    for (int kk = 0; kk < 6; ++kk)
      acc = wmma16(a2[kk], ldBT(ws_col, D_, nt, kk * 32, lane), acc);
    lg[nt] = acc;
  }
  #pragma unroll
  for (int r = 0; r < 8; ++r) {
    float mx = -1e30f;
    #pragma unroll
    for (int nt = 0; nt < 7; ++nt) {
      int c = nt * 16 + n16;
      float v = (c < C_) ? lg[nt][r] : -1e30f;
      lg[nt][r] = v;
      mx = fmaxf(mx, v);
    }
    #pragma unroll
    for (int o = 1; o < 16; o <<= 1) mx = fmaxf(mx, __shfl_xor(mx, o, 32));
    float s = 0.f;
    #pragma unroll
    for (int nt = 0; nt < 7; ++nt) {
      int c = nt * 16 + n16;
      float e = (c < C_) ? __expf(lg[nt][r] - mx) : 0.f;
      lg[nt][r] = e; s += e;
    }
    #pragma unroll
    for (int o = 1; o < 16; o <<= 1) s += __shfl_xor(s, o, 32);
    float inv = 1.f / s;
    int p = mt * 16 + r + ((lane & 16) ? 8 : 0);
    #pragma unroll
    for (int nt = 0; nt < 7; ++nt)
      maskL[p * S_MASK + nt * 16 + n16] = (_Float16)(lg[nt][r] * inv);
    maskL[p * S_MASK + 112 + n16] = (_Float16)0.f;   // zero K-pad cols 112..127
  }
  __syncthreads();

  // ---- x_emb[b] -> xeL transposed [d][c], parallel LN across all 256 threads ----
  // pass 1: partial sums, 2 threads per column (96 elements each)
  if (tid < 2 * C_) {
    int c = tid >> 1, h = tid & 1;
    float xv = x[(size_t)b * C_ + c];
    const float* wr = Wfe + c * D_ + h * (D_ / 2);
    const float* br = bfe + c * D_ + h * (D_ / 2);
    float s = 0.f, ss = 0.f;
    for (int d = 0; d < D_ / 2; ++d) {
      float t = xv * wr[d] + br[d]; t = t > 0.f ? t : 0.f;
      s += t; ss += t * t;
    }
    spart[tid] = s; sspart[tid] = ss;
  }
  __syncthreads();
  if (tid < C_) {
    float s  = spart[2 * tid] + spart[2 * tid + 1];
    float ss = sspart[2 * tid] + sspart[2 * tid + 1];
    float m = s / D_, var = ss / D_ - m * m;
    mrow[tid] = m;
    rsrow[tid] = rsqrtf(var + 1e-5f);
  }
  __syncthreads();
  // pass 2: all threads fill xeL[d][c] (c fastest -> contiguous LDS stores), pad c>=100 with 0
  for (int g = tid; g < 128 * D_; g += 256) {
    int d = g >> 7, c = g & 127;
    float val = 0.f;
    if (c < C_) {
      float xv = x[(size_t)b * C_ + c];
      float t = xv * Wfe[c * D_ + d] + bfe[c * D_ + d]; t = t > 0.f ? t : 0.f;
      val = (t - mrow[c]) * rsrow[c] * lnew[d] + lneb[d];
    }
    xeL[d * S_XE + c] = (_Float16)val;
  }
  __syncthreads();

  // ---- GEMM3: agg = mask @ x_emb (M=128,N=192,K=128); epilogue (softmax sum == 1) ----
  v16h a3[4];
  #pragma unroll
  for (int kk = 0; kk < 4; ++kk) a3[kk] = ldA(maskL, S_MASK, mt, kk * 32, lane);
  for (int nt = 0; nt < 12; ++nt) {
    v8f acc = {};
    #pragma unroll
    for (int kk = 0; kk < 4; ++kk)
      acc = wmma16(a3[kk], ldBT(xeL, S_XE, nt, kk * 32, lane), acc);
    int n = nt * 16 + n16;
    #pragma unroll
    for (int r = 0; r < 8; ++r) {
      int p = mr + r;
      out[((size_t)b * P_ + p) * D_ + n] = acc[r] * (1.f + Wexp[p]) + bexp[p];
    }
  }
}

// ---------------- launch ----------------

extern "C" void kernel_launch(void* const* d_in, const int* in_sizes, int n_in,
                              void* d_out, int out_size, void* d_ws, size_t ws_size,
                              hipStream_t stream) {
  (void)in_sizes; (void)n_in; (void)out_size; (void)ws_size;
  const float* x    = (const float*)d_in[0];
  const float* prev = (const float*)d_in[1];
  const float* Wfe  = (const float*)d_in[2];
  const float* bfe  = (const float*)d_in[3];
  const float* lnew = (const float*)d_in[4];
  const float* lneb = (const float*)d_in[5];
  const float* lncw = (const float*)d_in[6];
  const float* lncb = (const float*)d_in[7];
  const float* lnpw = (const float*)d_in[8];
  const float* lnpb = (const float*)d_in[9];
  const float* Wimp = (const float*)d_in[10];
  const float* bimp = (const float*)d_in[11];
  const float* embc = (const float*)d_in[12];
  const float* embp = (const float*)d_in[13];
  const float* Wexp = (const float*)d_in[14];
  const float* bexp = (const float*)d_in[15];
  float* out = (float*)d_out;

  char* ws = (char*)d_ws;
  _Float16* ws_col = (_Float16*)(ws + WS_COL);
  _Float16* ws_wt  = (_Float16*)(ws + WS_WT);
  float*    ws_pc  = (float*)(ws + WS_PC);

  k_colln<<<1, 128, 0, stream>>>(embc, lncw, lncb, ws_col);
  k_wt<<<(D_ * D_ + 255) / 256, 256, 0, stream>>>(Wimp, ws_wt);
  k_pconst<<<P_, D_, 0, stream>>>(embp, lnpw, lnpb, Wimp, bimp, ws_pc);
  k_main<<<B_, 256, 0, stream>>>(x, prev, Wfe, bfe, lnew, lneb,
                                 ws_col, ws_wt, ws_pc, Wexp, bexp, out);
}
